// AttentionHead_8229157339653
// MI455X (gfx1250) — matrix operand order repaired and checked by reading, not compiled, not run on hardware
//
#include <hip/hip_runtime.h>

// ---------------------------------------------------------------------------
// Single-head attention for MI455X (gfx1250), wave32 + WMMA f16->f32.
//   out = softmax( (xWq+bq)(xWk+bk)^T / sqrt(64) ) (xWv+bv)
// Pipeline:
//   K0: W -> f16 W^T  (contiguous B-fragment loads later)
//   K1: QKV projection, v_wmma_f32_16x16x32_f16, q pre-scaled by 1/8,
//       stores q,k row-major f16 and v transposed f16.
//   K2: flash attention; computes S^T = K*Q^T and O^T = V^T*P^T so softmax
//       rows map to lanes -> one shfl_xor(16) butterfly per reduction.
// ---------------------------------------------------------------------------

typedef __attribute__((ext_vector_type(16))) _Float16 v16h;
typedef __attribute__((ext_vector_type(8)))  _Float16 v8h;
typedef __attribute__((ext_vector_type(8)))  float    v8f;
typedef __attribute__((ext_vector_type(4)))  float    v4f;

#define EMB    1024
#define HEAD   64
#define NBATCH 4
#define SEQ    2048
#define NROWS  (NBATCH*SEQ)            // 8192 total rows

// workspace layout (units: _Float16 elements)
#define WS_Q  ((size_t)0)                    // [NROWS][HEAD]  q * (1/sqrt(64))
#define WS_K  ((size_t)NROWS*HEAD)           // [NROWS][HEAD]
#define WS_V  ((size_t)2*NROWS*HEAD)         // [HEAD][NROWS]  (transposed!)
#define WS_W  ((size_t)3*NROWS*HEAD)         // 3 x [HEAD][EMB] f16 W^T

static __device__ __forceinline__ v8f wmma_f16(v16h a, v16h b, v8f c) {
  // (neg_a, A, neg_b, B, c_mod, C, reuse_a, reuse_b)
  return __builtin_amdgcn_wmma_f32_16x16x32_f16(false, a, false, b,
                                                (short)0, c, false, false);
}

// A-fragment (16x32 f16, MxK) per-lane load. ISA layout: lane holds row
// M = lane&15; with kb = (lane>>4)*8 the 16 halves are K = kb..kb+7 then
// K = kb+16..kb+23. Caller passes p = row_ptr + kbase + (lane>>4)*8.
static __device__ __forceinline__ v16h load_a_frag(const _Float16* p) {
  v8h r0 = *(const v8h*)(p);
  v8h r1 = *(const v8h*)(p + 16);
  v16h a;
#pragma unroll
  for (int i = 0; i < 8; ++i) { a[i] = r0[i]; a[i + 8] = r1[i]; }
  return a;
}

// ---------------------------------------------------------------------------
// K0: transpose + f16-convert the three weight matrices.
// wt[m][n][e] = W_m[e][n]
// ---------------------------------------------------------------------------
__global__ void wt_transpose_kernel(const float* __restrict__ Wq,
                                    const float* __restrict__ Wk,
                                    const float* __restrict__ Wv,
                                    _Float16* __restrict__ wt) {
  int i = blockIdx.x * blockDim.x + threadIdx.x;
  if (i >= 3 * HEAD * EMB) return;
  int m = i / (HEAD * EMB);
  int r = i - m * (HEAD * EMB);
  int n = r / EMB;
  int e = r - n * EMB;
  const float* W = (m == 0) ? Wq : ((m == 1) ? Wk : Wv);
  wt[i] = (_Float16)W[e * HEAD + n];
}

// ---------------------------------------------------------------------------
// K1: QKV projection. One wave = 16 rows of x. 12 f32 accumulators
// (q,k,v x 4 N-tiles of 16). 12 WMMA per 32-wide K chunk, 32 chunks.
// ---------------------------------------------------------------------------
__global__ void __launch_bounds__(32) qkv_proj_kernel(
    const float* __restrict__ x, const _Float16* __restrict__ wt,
    const float* __restrict__ bq, const float* __restrict__ bk,
    const float* __restrict__ bv, _Float16* __restrict__ ws) {
  const int lane = threadIdx.x & 31;
  const int lo = lane & 15, hi = lane >> 4;
  const int rowbase = blockIdx.x * 16;

  const v8f vzero = {};
  v8f acc[3][4];
#pragma unroll
  for (int p = 0; p < 3; ++p)
#pragma unroll
    for (int t = 0; t < 4; ++t) acc[p][t] = vzero;

  // this lane's x row pointer, offset to its K-run (kb = hi*16)
  const float* xr = x + (size_t)(rowbase + lo) * EMB + hi * 16;

  for (int kc = 0; kc < EMB; kc += 32) {
    // A fragment: rows of x, K contiguous run [kb, kb+16)
    v4f r0a = *(const v4f*)(xr + kc);
    v4f r0b = *(const v4f*)(xr + kc + 4);
    v4f r1a = *(const v4f*)(xr + kc + 8);
    v4f r1b = *(const v4f*)(xr + kc + 12);
    v16h a;
#pragma unroll
    for (int i = 0; i < 4; ++i) {
      a[i]      = (_Float16)r0a[i];
      a[i + 4]  = (_Float16)r0b[i];
      a[i + 8]  = (_Float16)r1a[i];
      a[i + 12] = (_Float16)r1b[i];
    }
#pragma unroll
    for (int p = 0; p < 3; ++p) {
#pragma unroll
      for (int t = 0; t < 4; ++t) {
        // B fragment: lane = output col n = t*16+lo, K contiguous per half
        const _Float16* wp =
            wt + ((size_t)(p * HEAD + t * 16 + lo)) * EMB + kc + hi * 16;
        v16h b = *(const v16h*)wp;
        acc[p][t] = wmma_f16(a, b, acc[p][t]);
      }
    }
  }

  const float qscale = 0.125f;  // 1/sqrt(HEAD), folded into stored q
#pragma unroll
  for (int t = 0; t < 4; ++t) {
#pragma unroll
    for (int g = 0; g < 8; ++g) {
      const int col = t * 16 + lo;
      const int row = rowbase + g + hi * 8;  // C layout: row = g + 8*hi
      float qv = (acc[0][t][g] + bq[col]) * qscale;
      float kv = acc[1][t][g] + bk[col];
      float vv = acc[2][t][g] + bv[col];
      ws[WS_Q + (size_t)row * HEAD + col] = (_Float16)qv;
      ws[WS_K + (size_t)row * HEAD + col] = (_Float16)kv;
      ws[WS_V + (size_t)col * NROWS + row] = (_Float16)vv;  // transposed
    }
  }
}

// ---------------------------------------------------------------------------
// K2: flash attention. One wave = one 16-query tile of one batch.
// S^T = K*Q^T (queries on lanes), online softmax, O^T = V^T*P^T.
// ---------------------------------------------------------------------------
__global__ void __launch_bounds__(32) flash_attn_kernel(
    const _Float16* __restrict__ ws, float* __restrict__ out) {
  const _Float16* qm = ws + WS_Q;
  const _Float16* km = ws + WS_K;
  const _Float16* vt = ws + WS_V;
  const int lane = threadIdx.x & 31;
  const int lo = lane & 15, hi = lane >> 4;
  const int batch = blockIdx.x >> 7;   // 128 query tiles per batch
  const int qtile = blockIdx.x & 127;
  const int qrow0 = batch * SEQ + qtile * 16;
  const int kv0 = batch * SEQ;

  // Q^T B-fragments, kept in registers for the whole KV loop.
  // lane = query col lo, K (=dims) runs kb..kb+7, kb+16..kb+23 (kb = hi*8).
  v16h qb[2];
#pragma unroll
  for (int d = 0; d < 2; ++d)
    qb[d] = load_a_frag(qm + (size_t)(qrow0 + lo) * HEAD + d * 32 + hi * 8);

  const v8f vzero = {};
  v8f o[4];
#pragma unroll
  for (int t = 0; t < 4; ++t) o[t] = vzero;
  float m = -__builtin_inff();
  float l = 0.0f;

  for (int c = 0; c < SEQ; c += 32) {
    // ---- scores^T for 32 keys: s0 = keys c..c+15, s1 = keys c+16..c+31
    v8f s0 = vzero, s1 = vzero;
#pragma unroll
    for (int d = 0; d < 2; ++d) {
      v16h a0 = load_a_frag(km + (size_t)(kv0 + c + lo) * HEAD + d * 32 + hi * 8);
      v16h a1 = load_a_frag(km + (size_t)(kv0 + c + 16 + lo) * HEAD + d * 32 + hi * 8);
      s0 = wmma_f16(a0, qb[d], s0);
      s1 = wmma_f16(a1, qb[d], s1);
    }

    // ---- online softmax (per query = per lane column; combine halves)
    float cm = -__builtin_inff();
#pragma unroll
    for (int g = 0; g < 8; ++g) cm = fmaxf(cm, fmaxf(s0[g], s1[g]));
    cm = fmaxf(cm, __shfl_xor(cm, 16, 32));
    const float mnew = fmaxf(m, cm);
    const float corr = __expf(m - mnew);  // exp(-inf)=0 on first chunk

    float p0[8], p1[8];
    float cs = 0.0f;
#pragma unroll
    for (int g = 0; g < 8; ++g) {
      p0[g] = __expf(s0[g] - mnew);
      p1[g] = __expf(s1[g] - mnew);
      cs += p0[g] + p1[g];
    }
    cs += __shfl_xor(cs, 16, 32);
    l = l * corr + cs;
    m = mnew;

    // ---- assemble P^T B-fragment (32 keys x 16 queries) from C layout.
    // C holds (key = g + 8*hi + 16*frag, query = lo); B needs lane to hold
    // keys hi*8..hi*8+7 then 16+hi*8..16+hi*8+7 -> no shuffle needed.
    v16h pb;
#pragma unroll
    for (int g = 0; g < 8; ++g) {
      pb[g]     = (_Float16)p0[g];
      pb[g + 8] = (_Float16)p1[g];
    }

    // ---- O^T += V^T * P^T  (rescale accumulators first)
#pragma unroll
    for (int t = 0; t < 4; ++t) {
#pragma unroll
      for (int g = 0; g < 8; ++g) o[t][g] *= corr;
      v16h av = load_a_frag(vt + (size_t)(t * 16 + lo) * NROWS + kv0 + c + hi * 8);
      o[t] = wmma_f16(av, pb, o[t]);
    }
  }

  // ---- normalize and store: O^T frag holds (dim = t*16+g+8*hi, query = lo)
  const float inv = 1.0f / l;
#pragma unroll
  for (int t = 0; t < 4; ++t) {
#pragma unroll
    for (int g = 0; g < 8; ++g) {
      const int dim = t * 16 + g + hi * 8;
      out[(size_t)(qrow0 + lo) * HEAD + dim] = o[t][g] * inv;
    }
  }
}

// ---------------------------------------------------------------------------
extern "C" void kernel_launch(void* const* d_in, const int* in_sizes, int n_in,
                              void* d_out, int out_size, void* d_ws,
                              size_t ws_size, hipStream_t stream) {
  (void)in_sizes; (void)n_in; (void)out_size; (void)ws_size;
  const float* x  = (const float*)d_in[0];
  const float* Wq = (const float*)d_in[1];
  const float* bq = (const float*)d_in[2];
  const float* Wk = (const float*)d_in[3];
  const float* bk = (const float*)d_in[4];
  const float* Wv = (const float*)d_in[5];
  const float* bv = (const float*)d_in[6];
  _Float16* ws = (_Float16*)d_ws;
  float* out = (float*)d_out;

  // K0: 3 * 64 * 1024 elements
  wt_transpose_kernel<<<(3 * HEAD * EMB + 255) / 256, 256, 0, stream>>>(
      Wq, Wk, Wv, ws + WS_W);
  // K1: 8192 rows / 16 per wave = 512 single-wave workgroups
  qkv_proj_kernel<<<NROWS / 16, 32, 0, stream>>>(x, ws + WS_W, bq, bk, bv, ws);
  // K2: 4 batches * 128 query tiles = 512 single-wave workgroups
  flash_attn_kernel<<<NROWS / 16, 32, 0, stream>>>(ws, out);
}
